// GenericGraphEncoder_61177514164382
// MI455X (gfx1250) — matrix operand order repaired and checked by reading, not compiled, not run on hardware
//
#include <hip/hip_runtime.h>
#include <hip/hip_bf16.h>
#include <math.h>

typedef __attribute__((ext_vector_type(16))) __bf16 v16bf;
typedef __attribute__((ext_vector_type(8)))  float  v8f;
typedef __attribute__((ext_vector_type(4)))  unsigned int v4u;
typedef __attribute__((ext_vector_type(8)))  int    v8i;
typedef __attribute__((ext_vector_type(4)))  int    v4i;

#define THREADS 256
#define MAX_K   128          // max GEMM K (D_IN)

__device__ __forceinline__ __bf16 cvt_bf(float v)  { return (__bf16)v; }
__device__ __forceinline__ __bf16 cvt_bf(__bf16 v) { return v; }

// ---- WMMA A-fragment loaders (ISA 7.12.2 16-bit A 16x32 layout, wave32) ----
// lane = half*16 + m ; per-lane K values are two contiguous runs of 8:
//   run0: k0 + 8*half + [0..7]  -> a[0..7]
//   run1: k0 + 16 + 8*half + [0..7] -> a[8..15]
union bf16x8 { uint4 u; __bf16 h[8]; };

__device__ __forceinline__ v16bf load_a(const __bf16* __restrict__ A, int lda, int Nn,
                                        int m0, int k0, int lane) {
  int m = lane & 15, half = lane >> 4;
  int row = m0 + m; if (row >= Nn) row = Nn - 1;
  const __bf16* r = A + (size_t)row * lda + k0 + 8 * half;
  bf16x8 c0, c1;
  c0.u = *(const uint4*)(r);        // 16B vector load (global_load_b128)
  c1.u = *(const uint4*)(r + 16);
  v16bf a;
#pragma unroll
  for (int i = 0; i < 8; ++i) { a[i] = c0.h[i]; a[8 + i] = c1.h[i]; }
  return a;
}

__device__ __forceinline__ v16bf load_a(const float* __restrict__ A, int lda, int Nn,
                                        int m0, int k0, int lane) {
  int m = lane & 15, half = lane >> 4;
  int row = m0 + m; if (row >= Nn) row = Nn - 1;
  const float* r = A + (size_t)row * lda + k0 + 8 * half;
  float4 f0 = *(const float4*)(r);
  float4 f1 = *(const float4*)(r + 4);
  float4 f2 = *(const float4*)(r + 16);
  float4 f3 = *(const float4*)(r + 20);
  v16bf a;
  a[0] = cvt_bf(f0.x);  a[1] = cvt_bf(f0.y);  a[2]  = cvt_bf(f0.z);  a[3]  = cvt_bf(f0.w);
  a[4] = cvt_bf(f1.x);  a[5] = cvt_bf(f1.y);  a[6]  = cvt_bf(f1.z);  a[7]  = cvt_bf(f1.w);
  a[8] = cvt_bf(f2.x);  a[9] = cvt_bf(f2.y);  a[10] = cvt_bf(f2.z);  a[11] = cvt_bf(f2.w);
  a[12]= cvt_bf(f3.x);  a[13]= cvt_bf(f3.y);  a[14] = cvt_bf(f3.z);  a[15] = cvt_bf(f3.w);
  return a;
}

// ---- GEMM: H[N,64] = A[N,Kt] @ W[Kt,64]  (bf16 WMMA, f32 accumulate) -------
// Weight tile staged into LDS once per workgroup via the Tensor Data Mover;
// one wave computes a 16-row strip x four 16-col tiles.
template <typename AT>
__global__ void k_gemm(const AT* __restrict__ A, const float* __restrict__ W,
                       float* __restrict__ H, int Nn, int Kt) {
  __shared__ float sW[MAX_K * 64];      // sole LDS object -> LDS offset 0

#if defined(__gfx1250__) && __has_builtin(__builtin_amdgcn_tensor_load_to_lds) && \
    __has_builtin(__builtin_amdgcn_s_wait_tensorcnt)
  if (threadIdx.x < 32) {               // wave 0 issues the TDM transfer
    unsigned long long ga = (unsigned long long)W;
    v4u g0;
    g0[0] = 1u;                                          // count=1 (valid D#)
    g0[1] = 0u;                                          // lds_addr = 0
    g0[2] = (unsigned)(ga & 0xffffffffu);                // global_addr[31:0]
    g0[3] = (unsigned)((ga >> 32) & 0x01ffffffu)         // global_addr[56:32]
            | 0x80000000u;                               // type = 2 ("image")
    v8i g1;
    g1[0] = (int)(2u << 16);                             // data_size = 4B
    g1[1] = (int)(64u << 16);                            // tensor_dim0 = 64
    g1[2] = (int)((unsigned)Kt << 16);                   // tensor_dim1 = Kt
    g1[3] = (int)(64u << 16);                            // tile_dim0 = 64
    g1[4] = (int)(unsigned)Kt;                           // tile_dim1 = Kt
    g1[5] = 64;                                          // tensor_dim0_stride = 64
    g1[6] = 0;
    g1[7] = 0;
    v4i g2 = {0, 0, 0, 0}, g3 = {0, 0, 0, 0};            // 2D tensor: groups 2/3 unused
    v8i g4 = {0, 0, 0, 0, 0, 0, 0, 0};                   // extra group (zero-fill)
    __builtin_amdgcn_tensor_load_to_lds(g0, g1, g2, g3, g4, 0);
    __builtin_amdgcn_s_wait_tensorcnt(0);
  }
  __syncthreads();
#else
  for (int i = threadIdx.x; i < Kt * 64; i += blockDim.x) sW[i] = W[i];
  __syncthreads();
#endif

  int wave = (int)((blockIdx.x * blockDim.x + threadIdx.x) >> 5);
  int lane = threadIdx.x & 31;
  int strips = (Nn + 15) >> 4;
  if (wave >= strips) return;           // wave-uniform: EXEC stays all-ones
  int m0 = wave << 4;

  v8f acc[4];
#pragma unroll
  for (int t = 0; t < 4; ++t)
#pragma unroll
    for (int r = 0; r < 8; ++r) acc[t][r] = 0.0f;

  // B frag: col n = lane%16, K base = 8*(lane/16), VGPR v packs
  // K = kb+2v(+8 if v>=4), +1 (matches A). Read f32 from LDS, convert inline.
  int bn = lane & 15, bkb = (lane >> 4) * 8;

  for (int ks = 0; ks < Kt; ks += 32) {
    v16bf a = load_a(A, Kt, Nn, m0, ks, lane);
#pragma unroll
    for (int t = 0; t < 4; ++t) {
      v16bf b;
#pragma unroll
      for (int v = 0; v < 8; ++v) {
        int k = ks + bkb + 2 * v + (v >= 4 ? 8 : 0);
        b[2 * v]     = cvt_bf(sW[k * 64 + t * 16 + bn]);
        b[2 * v + 1] = cvt_bf(sW[(k + 1) * 64 + t * 16 + bn]);
      }
      acc[t] = __builtin_amdgcn_wmma_f32_16x16x32_bf16(
          false, a, false, b, (short)0, acc[t], false, false);
    }
  }
  // C/D layout: VGPR r -> M = m0 + 8*(lane/16) + r, N = lane%16
  int n = lane & 15, mo = (lane >> 4) * 8;
#pragma unroll
  for (int t = 0; t < 4; ++t)
#pragma unroll
    for (int r = 0; r < 8; ++r) {
      int row = m0 + mo + r;
      if (row < Nn) H[(size_t)row * 64 + t * 16 + n] = acc[t][r];
    }
}

// ---- degree / edge-norm precompute -----------------------------------------
__global__ void k_fill_f32(float* p, float v, int n) {
  int i = blockIdx.x * blockDim.x + threadIdx.x;
  if (i < n) p[i] = v;
}
__global__ void k_deg_count(const int* __restrict__ dst, float* deg, int E) {
  int e = blockIdx.x * blockDim.x + threadIdx.x;
  if (e < E) atomicAdd(&deg[dst[e]], 1.0f);
}
__global__ void k_dinv(float* deg, int N) {
  int i = blockIdx.x * blockDim.x + threadIdx.x;
  if (i < N) deg[i] = rsqrtf(deg[i]);
}
__global__ void k_edge_w(const int* __restrict__ src, const int* __restrict__ dst,
                         const float* __restrict__ dinv, float* __restrict__ ew, int E) {
  int e = blockIdx.x * blockDim.x + threadIdx.x;
  if (e < E) ew[e] = dinv[src[e]] * dinv[dst[e]];
}

// ---- out = h*self_w + b  (written into x at column offset) -----------------
__global__ void k_self_bias(const float* __restrict__ H, const float* __restrict__ dinv,
                            const float* __restrict__ bias, float* __restrict__ X,
                            int N, int C, int off) {
  int i = blockIdx.x * blockDim.x + threadIdx.x;
  if (i >= N * 64) return;
  int n = i >> 6, c = i & 63;
  float dv = dinv[n];
  X[(size_t)n * C + off + c] = H[i] * dv * dv + bias[c];
}

// ---- edge scatter: x[dst, off+c] += h[src, c] * ew[e]  (wave per edge) -----
__global__ void k_scatter(const float* __restrict__ H, const float* __restrict__ ew,
                          const int* __restrict__ src, const int* __restrict__ dst,
                          float* __restrict__ X, int E, int C, int off) {
  int wave = (int)((blockIdx.x * blockDim.x + threadIdx.x) >> 5);
  int lane = threadIdx.x & 31;
  if (wave >= E) return;
  int s = src[wave], d = dst[wave];
  float w = ew[wave];
  const float* hs = H + (size_t)s * 64;
  float* o = X + (size_t)d * C + off;
  atomicAdd(&o[lane],      hs[lane]      * w);
  atomicAdd(&o[lane + 32], hs[lane + 32] * w);
}

// ---- graph-LayerNorm: global mean/var over all N*64 elements ---------------
__global__ void k_zero2(float* red) { red[0] = 0.0f; red[1] = 0.0f; }

__global__ void k_reduce(const float* __restrict__ X, int N, int C, int off,
                         float* __restrict__ red) {
  __shared__ float ssum[THREADS], ssq[THREADS];
  float s = 0.f, q = 0.f;
  size_t total = (size_t)N * 64;
  for (size_t i = (size_t)blockIdx.x * blockDim.x + threadIdx.x; i < total;
       i += (size_t)gridDim.x * blockDim.x) {
    int n = (int)(i >> 6), c = (int)(i & 63);
    float v = X[(size_t)n * C + off + c];
    s += v; q += v * v;
  }
  ssum[threadIdx.x] = s; ssq[threadIdx.x] = q;
  __syncthreads();
  for (int st = THREADS / 2; st > 0; st >>= 1) {
    if ((int)threadIdx.x < st) {
      ssum[threadIdx.x] += ssum[threadIdx.x + st];
      ssq[threadIdx.x]  += ssq[threadIdx.x + st];
    }
    __syncthreads();
  }
  if (threadIdx.x == 0) { atomicAdd(&red[0], ssum[0]); atomicAdd(&red[1], ssq[0]); }
}

__global__ void k_ln_act(const float* __restrict__ X, int N, int C, int off,
                         const float* __restrict__ lnw, const float* __restrict__ lnb,
                         const float* __restrict__ red, __bf16* __restrict__ Hin) {
  int i = blockIdx.x * blockDim.x + threadIdx.x;
  if (i >= N * 64) return;
  int n = i >> 6, c = i & 63;
  float cnt = (float)N * 64.0f;
  float m   = red[0] / cnt;
  float var = red[1] / cnt - m * m;
  float sd  = sqrtf(fmaxf(var, 0.0f));
  float v = (X[(size_t)n * C + off + c] - m) / (sd + 1e-5f) * lnw[c] + lnb[c];
  v = (v > 0.0f) ? v : 0.01f * v;           // leaky_relu
  Hin[i] = (__bf16)v;
}

// ---- per-graph ranges (batch_index is sorted) ------------------------------
__global__ void k_gb_init(int* gs, int* ge, int G) {
  int g = blockIdx.x * blockDim.x + threadIdx.x;
  if (g < G) { gs[g] = 0; ge[g] = 0; }
}
__global__ void k_gbounds(const int* __restrict__ batch, int N, int* gs, int* ge) {
  int n = blockIdx.x * blockDim.x + threadIdx.x;
  if (n >= N) return;
  int b = batch[n];
  if (n == 0     || batch[n - 1] != b) gs[b] = n;
  if (n == N - 1 || batch[n + 1] != b) ge[b] = n + 1;
}

// ---- softmax aggregation per graph, per channel ----------------------------
__global__ void k_smax(const float* __restrict__ X, const int* __restrict__ gs,
                       const int* __restrict__ ge, const float* __restrict__ tptr,
                       float* __restrict__ reps, int C) {
  int g = blockIdx.x;
  int s = gs[g], e = ge[g];
  float t = tptr[0];
  for (int base = 0; base < C; base += blockDim.x) {
    int c = base + threadIdx.x;
    if (c >= C) continue;
    float m = -3.402823466e38f;
    for (int n = s; n < e; ++n) m = fmaxf(m, X[(size_t)n * C + c] * t);
    float se = 0.f, sw = 0.f;
    for (int n = s; n < e; ++n) {
      float xv = X[(size_t)n * C + c];
      float ey = expf(xv * t - m);
      se += ey; sw += xv * ey;
    }
    reps[(size_t)g * C + c] = (e > s) ? (sw / se) : 0.0f;
  }
}

// ---------------------------------------------------------------------------
extern "C" void kernel_launch(void* const* d_in, const int* in_sizes, int n_in,
                              void* d_out, int out_size, void* d_ws, size_t ws_size,
                              hipStream_t stream) {
  const float* Xin  = (const float*)d_in[0];
  const int*   ei   = (const int*)d_in[1];
  const int*   batch= (const int*)d_in[3];
  const float* W0   = (const float*)d_in[4];
  const float* b0   = (const float*)d_in[5];
  const float* lnw  = (const float*)d_in[6];
  const float* lnb  = (const float*)d_in[7];
  const float* Ws   = (const float*)d_in[8];
  const float* bs   = (const float*)d_in[9];
  const float* tp   = (const float*)d_in[10];

  const int N   = in_sizes[3];
  const int E   = in_sizes[2];
  const int DIN = in_sizes[0] / N;          // 128
  const int DH  = in_sizes[5];              // 64
  const int L   = in_sizes[6] / DH;         // 12
  const int C   = (L + 1) * DH;             // 832
  const int G   = out_size / C - N;         // 64

  float* reps = (float*)d_out;
  float* x    = reps + (size_t)G * C;

  const int* src = ei;
  const int* dst = ei + E;

  // workspace layout (256B aligned slices)
  auto aup = [](size_t v) { return (v + 255) & ~(size_t)255; };
  char* w = (char*)d_ws;
  float*  dinv = (float*)w;  w += aup((size_t)N * 4);
  float*  ew   = (float*)w;  w += aup((size_t)E * 4);
  float*  h    = (float*)w;  w += aup((size_t)N * DH * 4);
  __bf16* hin  = (__bf16*)w; w += aup((size_t)N * DH * 2);
  float*  red  = (float*)w;  w += 256;
  int*    gsb  = (int*)w;    w += aup((size_t)G * 4);
  int*    geb  = (int*)w;    w += aup((size_t)G * 4);
  (void)ws_size; (void)n_in;

  const int gN   = (N + THREADS - 1) / THREADS;
  const int gE   = (E + THREADS - 1) / THREADS;
  const int gNE  = (N * 64 + THREADS - 1) / THREADS;
  const int gSc  = (E * 32 + THREADS - 1) / THREADS;       // wave per edge
  const int strips = (N + 15) >> 4;
  const int gGm  = (strips * 32 + THREADS - 1) / THREADS;  // wave per 16-row strip
  const int gG   = (G + THREADS - 1) / THREADS;

  // degree / symmetric normalization
  k_fill_f32 <<<gN, THREADS, 0, stream>>>(dinv, 1.0f, N);  // deg = 1 (self-loop)
  k_deg_count<<<gE, THREADS, 0, stream>>>(dst, dinv, E);
  k_dinv     <<<gN, THREADS, 0, stream>>>(dinv, N);        // deg^-0.5 in place
  k_edge_w   <<<gE, THREADS, 0, stream>>>(src, dst, dinv, ew, E);

  // layer 0: h = X @ W0 ; out0 = scatter + self + bias -> x[:, 0:64]
  k_gemm<float><<<gGm, THREADS, 0, stream>>>(Xin, W0, h, N, DIN);
  k_self_bias  <<<gNE, THREADS, 0, stream>>>(h, dinv, b0, x, N, C, 0);
  k_scatter    <<<gSc, THREADS, 0, stream>>>(h, ew, src, dst, x, E, C, 0);

  for (int i = 0; i < L; ++i) {
    const int off  = i * DH;
    const int offo = (i + 1) * DH;
    k_zero2  <<<1, 1, 0, stream>>>(red);
    k_reduce <<<2048, THREADS, 0, stream>>>(x, N, C, off, red);
    k_ln_act <<<gNE, THREADS, 0, stream>>>(x, N, C, off, lnw + (size_t)i * DH,
                                           lnb + (size_t)i * DH, red, hin);
    k_gemm<__bf16><<<gGm, THREADS, 0, stream>>>(hin, Ws + (size_t)i * DH * DH, h, N, DH);
    k_self_bias   <<<gNE, THREADS, 0, stream>>>(h, dinv, bs + (size_t)i * DH, x, N, C, offo);
    k_scatter     <<<gSc, THREADS, 0, stream>>>(h, ew, src, dst, x, E, C, offo);
  }

  // per-graph softmax aggregation
  k_gb_init<<<gG, THREADS, 0, stream>>>(gsb, geb, G);
  k_gbounds<<<gN, THREADS, 0, stream>>>(batch, N, gsb, geb);
  k_smax   <<<G, THREADS, 0, stream>>>(x, gsb, geb, tp, reps, C);
}
